// EEGMamba_59622736003861
// MI455X (gfx1250) — matrix operand order repaired and checked by reading, not compiled, not run on hardware
//
#include <hip/hip_runtime.h>
#include <hip/hip_bf16.h>

typedef __attribute__((ext_vector_type(16))) _Float16 v16h;
typedef __attribute__((ext_vector_type(8)))  _Float16 v8h;
typedef __attribute__((ext_vector_type(8)))  float    v8f;
typedef __attribute__((ext_vector_type(4)))  float    v4f;

#define Bsz 64
#define Tlen 1024
#define Dm 256
#define DI 512
#define Mrows 65536   // B*T

// ---------------------------------------------------------------------------
// Initial projection: h[b*T+t][d] = bp[d] + sum_c x[b][c][t] * Wp[c][d]
// ---------------------------------------------------------------------------
__global__ __launch_bounds__(256)
void inproj_kernel(const float* __restrict__ x, const float* __restrict__ Wp,
                   const float* __restrict__ bp, float* __restrict__ h)
{
    size_t row = blockIdx.x;          // 0..65535
    int dcol = threadIdx.x;           // 0..255
    int b = (int)(row >> 10);
    int t = (int)(row & 1023);
    float acc = bp[dcol];
#pragma unroll
    for (int c = 0; c < 16; ++c)
        acc += x[((size_t)b * 16 + c) * 1024 + t] * Wp[c * 256 + dcol];
    h[row * 256 + dcol] = acc;
}

// ---------------------------------------------------------------------------
// Weight convert+transpose: WT[n][k] (f16) = W[k][n] (f32). One-shot per launch.
// ---------------------------------------------------------------------------
__global__ __launch_bounds__(256)
void wcvt_kernel(const float* __restrict__ W, _Float16* __restrict__ WT, int K, int N)
{
    int idx = blockIdx.x * 256 + threadIdx.x;
    if (idx >= K * N) return;
    int k = idx / N, n = idx - k * N;          // n fastest -> coalesced reads
    WT[(size_t)n * K + k] = (_Float16)W[idx];
}

// ---------------------------------------------------------------------------
// LayerNorm over D=256 per row, f16 output
// ---------------------------------------------------------------------------
__global__ __launch_bounds__(256)
void ln_kernel(const float* __restrict__ h, const float* __restrict__ g,
               const float* __restrict__ bta, _Float16* __restrict__ hn)
{
    __shared__ float sb[256];
    size_t row = blockIdx.x;
    int d = threadIdx.x;
    float v = h[row * 256 + d];
    sb[d] = v; __syncthreads();
    for (int s = 128; s > 0; s >>= 1) { if (d < s) sb[d] += sb[d + s]; __syncthreads(); }
    float mean = sb[0] * (1.f / 256.f);
    __syncthreads();
    float c = v - mean;
    sb[d] = c * c; __syncthreads();
    for (int s = 128; s > 0; s >>= 1) { if (d < s) sb[d] += sb[d + s]; __syncthreads(); }
    float var = sb[0] * (1.f / 256.f);
    float r = rsqrtf(var + 1e-5f);
    hn[row * 256 + d] = (_Float16)(c * r * g[d] + bta[d]);
}

// ---------------------------------------------------------------------------
// Templated WMMA GEMM with LDS double buffering.
//  out = act( A[f16, M x KT, stride LDA] @ WT^T + bias ), WT = f16 [NT][KT].
//  128 threads = 4 waves; each wave computes a 32x32 subtile of a 64x64 tile.
//  One barrier per K-step; next tile's global loads issued before compute.
// ---------------------------------------------------------------------------
template<int KT, int LDA, int NT, int ACT, bool HASBIAS, bool ACCUM>
__global__ __launch_bounds__(128, 1)
void gemm_wmma(const _Float16* __restrict__ A, const _Float16* __restrict__ WT,
               const float* __restrict__ bias,
               _Float16* __restrict__ outH, float* __restrict__ outF,
               int revA, int revC)
{
    __shared__ __align__(16) _Float16 smem[10240];   // 20 KB: 2 x (As 64x40 + Bs 64x40)

    const int tid  = threadIdx.x;
    const int wave = tid >> 5, lane = tid & 31;
    const int m0 = blockIdx.x * 64;
    const int n0 = blockIdx.y * 64;
    const int wm = (wave >> 1) * 32;
    const int wn = (wave &  1) * 32;

    v8f acc[2][2] = {};

    // staging assignment: 2 threads per row, 16 halves each (A and B identical)
    const int sr = tid >> 1;
    const int sc = (tid & 1) * 16;
    int gm = m0 + sr;
    if (revA) { int bb = gm >> 10; int tt = gm & 1023; gm = (bb << 10) | (1023 - tt); }
    const _Float16* Arow = A + (size_t)gm * LDA;
    int nrow = n0 + sr;
    if constexpr (NT % 64 != 0) nrow = (nrow < NT) ? nrow : NT - 1;  // clamp; cols >=NT discarded
    const _Float16* Brow = WT + (size_t)nrow * KT;

    const int soff = sr * 40 + sc;

    // ---- prologue: stage tile 0 into buffer 0 ----
    {
        _Float16* bA = smem;
        _Float16* bB = smem + 2560;
        if constexpr (KT % 32 == 0) {
            v8h a0 = *(const v8h*)(Arow + sc);
            v8h a1 = *(const v8h*)(Arow + sc + 8);
            v8h b0 = *(const v8h*)(Brow + sc);
            v8h b1 = *(const v8h*)(Brow + sc + 8);
            *(v8h*)&bA[soff]     = a0;
            *(v8h*)&bA[soff + 8] = a1;
            *(v8h*)&bB[soff]     = b0;
            *(v8h*)&bB[soff + 8] = b1;
        } else {
            // KT == 16: halves 16..31 are zero padding
            if (sc == 0) {
                v8h a0 = *(const v8h*)(Arow + 0);
                v8h a1 = *(const v8h*)(Arow + 8);
                v8h b0 = *(const v8h*)(Brow + 0);
                v8h b1 = *(const v8h*)(Brow + 8);
                *(v8h*)&bA[soff]     = a0;
                *(v8h*)&bA[soff + 8] = a1;
                *(v8h*)&bB[soff]     = b0;
                *(v8h*)&bB[soff + 8] = b1;
            } else {
                v8h z = {};
                *(v8h*)&bA[soff]     = z;
                *(v8h*)&bA[soff + 8] = z;
                *(v8h*)&bB[soff]     = z;
                *(v8h*)&bB[soff + 8] = z;
            }
        }
    }
    __syncthreads();

    constexpr int KTILES = (KT + 31) / 32;
    const int fl = lane & 15;
    const int ah = (lane >> 4) * 8;    // A: elems 0..7 = K ah..+7 ; 8..15 = K ah+16..+23
    const int bh = (lane >> 4) * 8;    // B: elems 0..7 = K bh..+7 ; 8..15 = K bh+16..+23

    for (int kt = 0; kt < KTILES; ++kt) {
        const _Float16* cA = smem + (kt & 1) * 5120;
        const _Float16* cB = cA + 2560;

        // issue next tile's global loads first (overlap with WMMA)
        v8h na0, na1, nb0, nb1;
        bool hasNext = false;
        if constexpr (KT % 32 == 0) {
            hasNext = (kt + 1 < KTILES);
            if (hasNext) {
                const int kk = (kt + 1) * 32;
                na0 = *(const v8h*)(Arow + kk + sc);
                na1 = *(const v8h*)(Arow + kk + sc + 8);
                nb0 = *(const v8h*)(Brow + kk + sc);
                nb1 = *(const v8h*)(Brow + kk + sc + 8);
                if (kk + 32 < KT) {
                    __builtin_prefetch(Arow + kk + 32 + sc, 0, 1);
                    __builtin_prefetch(Brow + kk + 32 + sc, 0, 1);
                }
            }
        }

        // ---- fragments per ISA layouts ----
        v16h afrag[2], bfrag[2];
#pragma unroll
        for (int i = 0; i < 2; ++i) {
            v8h lo = *(const v8h*)&cA[(wm + i * 16 + fl) * 40 + ah];
            v8h hi = *(const v8h*)&cA[(wm + i * 16 + fl) * 40 + ah + 16];
#pragma unroll
            for (int e = 0; e < 8; ++e) { afrag[i][e] = lo[e]; afrag[i][e + 8] = hi[e]; }
        }
#pragma unroll
        for (int j = 0; j < 2; ++j) {
            v8h lo = *(const v8h*)&cB[(wn + j * 16 + fl) * 40 + bh];
            v8h hi = *(const v8h*)&cB[(wn + j * 16 + fl) * 40 + bh + 16];
#pragma unroll
            for (int e = 0; e < 8; ++e) { bfrag[j][e] = lo[e]; bfrag[j][e + 8] = hi[e]; }
        }
#pragma unroll
        for (int i = 0; i < 2; ++i)
#pragma unroll
            for (int j = 0; j < 2; ++j)
                acc[i][j] = __builtin_amdgcn_wmma_f32_16x16x32_f16(
                    false, afrag[i], false, bfrag[j], (short)0, acc[i][j], false, false);

        // store next tile into alternate buffer
        if constexpr (KT % 32 == 0) {
            if (hasNext) {
                _Float16* nA = smem + ((kt + 1) & 1) * 5120;
                _Float16* nB = nA + 2560;
                *(v8h*)&nA[soff]     = na0;
                *(v8h*)&nA[soff + 8] = na1;
                *(v8h*)&nB[soff]     = nb0;
                *(v8h*)&nB[soff + 8] = nb1;
            }
        }
        __syncthreads();
    }

    const int col = lane & 15;
    const int mb  = (lane >> 4) * 8;

    if constexpr (!ACCUM) {
        // ---- f16 store epilogue: swizzle 64x64 tile through LDS ----
        _Float16* ct = smem;                       // stride 72 (64*72=4608)
#pragma unroll
        for (int j = 0; j < 2; ++j) {
            float bv = 0.f;
            if constexpr (HASBIAS) bv = bias[n0 + wn + j * 16 + col];
#pragma unroll
            for (int i = 0; i < 2; ++i)
#pragma unroll
                for (int r = 0; r < 8; ++r) {
                    float v = acc[i][j][r];
                    if constexpr (HASBIAS) v += bv;
                    if constexpr (ACT == 1) v = (v > 20.f) ? v : log1pf(__expf(v));
                    ct[(wm + i * 16 + mb + r) * 72 + wn + j * 16 + col] = (_Float16)v;
                }
        }
        __syncthreads();
        const int row = tid >> 1;
        const int ch  = (tid & 1) * 32;
        const size_t m = (size_t)(m0 + row);
#pragma unroll
        for (int u = 0; u < 4; ++u) {
            int n = n0 + ch + u * 8;
            bool ok = true;
            if constexpr (NT % 64 != 0) ok = (n + 8) <= NT;
            if (ok)
                *(v8h*)&outH[m * NT + n] = *(const v8h*)&ct[row * 72 + ch + u * 8];
        }
    } else {
        // ---- f32 accumulate epilogue: two 32x64 half-tiles through LDS ----
        float* cf = (float*)smem;                  // stride 68 (32*68=2176)
#pragma unroll
        for (int half = 0; half < 2; ++half) {
            __syncthreads();
            if (wm == half * 32) {
#pragma unroll
                for (int i = 0; i < 2; ++i)
#pragma unroll
                    for (int j = 0; j < 2; ++j)
#pragma unroll
                        for (int r = 0; r < 8; ++r)
                            cf[(i * 16 + mb + r) * 68 + wn + j * 16 + col] = acc[i][j][r];
            }
            __syncthreads();
            const int row = tid >> 2;              // 0..31
            const int c0  = (tid & 3) * 16;        // 0,16,32,48
            int m = m0 + half * 32 + row;
            int om = m;
            if (revC) { int bb = m >> 10; int tt = m & 1023; om = (bb << 10) | (1023 - tt); }
            float* op = outF + (size_t)om * NT + n0 + c0;
            const float* cp = cf + row * 68 + c0;
#pragma unroll
            for (int u = 0; u < 4; ++u) {
                v4f o = *(v4f*)(op + u * 4);
                v4f c = *(const v4f*)(cp + u * 4);
                o += c;
                *(v4f*)(op + u * 4) = o;
            }
        }
    }
}

// ---------------------------------------------------------------------------
// Depthwise causal conv (K=4) + SiLU on xc = xz[:, :512] -> xs
// ---------------------------------------------------------------------------
__global__ __launch_bounds__(256)
void conv_silu_kernel(const _Float16* __restrict__ xz, const float* __restrict__ convw,
                      const float* __restrict__ convb, _Float16* __restrict__ xs)
{
    size_t idx = (size_t)blockIdx.x * 256 + threadIdx.x;   // M*512
    int d = (int)(idx & 511);
    size_t row = idx >> 9;
    int t = (int)(row & 1023);
    float acc = convb[d];
#pragma unroll
    for (int k = 0; k < 4; ++k) {
        int tt = t + k - 3;
        if (tt >= 0)
            acc += (float)xz[(row + (size_t)(k - 3)) * 1024 + d] * convw[d * 4 + k];
    }
    float s = acc / (1.f + __expf(-acc));   // silu
    xs[row * 512 + d] = (_Float16)s;
}

// ---------------------------------------------------------------------------
// Selective scan: one lane per (b,d); S=16 state in VGPRs, sequential over T.
// Fuses y' = (y + xs*Dp) * silu(z) epilogue. Writes yz (f16).
// ---------------------------------------------------------------------------
__global__ __launch_bounds__(256)
void scan_kernel(const _Float16* __restrict__ dt, const _Float16* __restrict__ xs,
                 const _Float16* __restrict__ pj, const _Float16* __restrict__ xz,
                 const float* __restrict__ Alog, const float* __restrict__ Dp,
                 _Float16* __restrict__ yz)
{
    int idx = blockIdx.x * 256 + threadIdx.x;   // 0..32767
    int d = idx & 511;
    int b = idx >> 9;
    float A[16];
#pragma unroll
    for (int s = 0; s < 16; ++s) A[s] = -__expf(Alog[d * 16 + s]);
    float Dv = Dp[d];
    float hs[16];
#pragma unroll
    for (int s = 0; s < 16; ++s) hs[s] = 0.f;

    for (int t = 0; t < 1024; ++t) {
        size_t row = (size_t)b * 1024 + t;
        float dtv = (float)dt[row * 512 + d];
        float xv  = (float)xs[row * 512 + d];
        const _Float16* pr = pj + row * 48;
        float dx = dtv * xv;
        float y = 0.f;
#pragma unroll
        for (int s = 0; s < 16; ++s) {
            float dA = __expf(dtv * A[s]);
            hs[s] = dA * hs[s] + dx * (float)pr[16 + s];   // Bm
            y += hs[s] * (float)pr[32 + s];                // Cm
        }
        float zv = (float)xz[row * 1024 + 512 + d];
        float out = (y + xv * Dv) * (zv / (1.f + __expf(-zv)));
        yz[row * 512 + d] = (_Float16)out;
    }
}

// ---------------------------------------------------------------------------
// Mean-pool over T
// ---------------------------------------------------------------------------
__global__ __launch_bounds__(256)
void pool_kernel(const float* __restrict__ h, float* __restrict__ pooled)
{
    int b = blockIdx.x; int d = threadIdx.x;
    float s = 0.f;
    for (int t = 0; t < 1024; ++t)
        s += h[((size_t)b * 1024 + t) * 256 + d];
    pooled[b * 256 + d] = s * (1.f / 1024.f);
}

// ---------------------------------------------------------------------------
// Head: LN -> gelu(p@Wc1+bc1) -> @Wc2 + bc2   (tiny: one block per batch row)
// ---------------------------------------------------------------------------
__global__ __launch_bounds__(128)
void head_kernel(const float* __restrict__ pooled, const float* __restrict__ gg,
                 const float* __restrict__ bb, const float* __restrict__ Wc1,
                 const float* __restrict__ bc1, const float* __restrict__ Wc2,
                 const float* __restrict__ bc2, float* __restrict__ out)
{
    __shared__ float pn[256];
    __shared__ float red[128];
    __shared__ float mv[2];
    int b = blockIdx.x, j = threadIdx.x;
    if (j == 0) {
        float s = 0.f;
        for (int d = 0; d < 256; ++d) s += pooled[b * 256 + d];
        float m = s / 256.f, q = 0.f;
        for (int d = 0; d < 256; ++d) { float c = pooled[b * 256 + d] - m; q += c * c; }
        mv[0] = m; mv[1] = q / 256.f;
    }
    __syncthreads();
    float m = mv[0], r = rsqrtf(mv[1] + 1e-5f);
    for (int d = j; d < 256; d += 128)
        pn[d] = (pooled[b * 256 + d] - m) * r * gg[d] + bb[d];
    __syncthreads();
    float acc = bc1[j];
    for (int d = 0; d < 256; ++d) acc += pn[d] * Wc1[d * 128 + j];
    float u = 0.7978845608028654f * (acc + 0.044715f * acc * acc * acc);
    float ge = 0.5f * acc * (1.f + tanhf(u));               // gelu (tanh approx)
    red[j] = ge * Wc2[j];
    __syncthreads();
    for (int s = 64; s > 0; s >>= 1) { if (j < s) red[j] += red[j + s]; __syncthreads(); }
    if (j == 0) out[b] = red[0] + bc2[0];
}

// ---------------------------------------------------------------------------
extern "C" void kernel_launch(void* const* d_in, const int* in_sizes, int n_in,
                              void* d_out, int out_size, void* d_ws, size_t ws_size,
                              hipStream_t stream)
{
    (void)in_sizes; (void)n_in; (void)out_size; (void)ws_size;

    const float* x     = (const float*)d_in[0];
    const float* Wp    = (const float*)d_in[1];
    const float* bp    = (const float*)d_in[2];
    const float* ln_g  = (const float*)d_in[3];
    const float* ln_b  = (const float*)d_in[4];
    const float* Win   = (const float*)d_in[5];
    const float* convw = (const float*)d_in[6];
    const float* convb = (const float*)d_in[7];
    const float* Wx    = (const float*)d_in[8];
    const float* Wdt   = (const float*)d_in[9];
    const float* bdt   = (const float*)d_in[10];
    const float* Alog  = (const float*)d_in[11];
    const float* Dp    = (const float*)d_in[12];
    const float* Wout  = (const float*)d_in[13];
    const float* lnf_g = (const float*)d_in[14];
    const float* lnf_b = (const float*)d_in[15];
    const float* Wc1   = (const float*)d_in[16];
    const float* bc1   = (const float*)d_in[17];
    const float* Wc2   = (const float*)d_in[18];
    const float* bc2   = (const float*)d_in[19];

    // workspace layout
    char* w = (char*)d_ws;
    float*    h_buf  = (float*)w;    w += (size_t)Mrows * 256 * 4;   // 64 MB
    _Float16* hn_buf = (_Float16*)w; w += (size_t)Mrows * 256 * 2;   // 32 MB
    _Float16* xz_buf = (_Float16*)w; w += (size_t)Mrows * 1024 * 2;  // 128 MB
    _Float16* xs_buf = (_Float16*)w; w += (size_t)Mrows * 512 * 2;   // 64 MB
    _Float16* pj_buf = (_Float16*)w; w += (size_t)Mrows * 48 * 2;    // 6 MB
    _Float16* dt_buf = (_Float16*)w; w += (size_t)Mrows * 512 * 2;   // 64 MB
    _Float16* yz_buf = (_Float16*)w; w += (size_t)Mrows * 512 * 2;   // 64 MB
    float*    pooled = (float*)w;    w += (size_t)64 * 256 * 4;
    // pre-transposed f16 weights (per layer-direction)
    _Float16* wtWin  = (_Float16*)w; w += (size_t)8 * 1024 * 256 * 2;  // 4 MB
    _Float16* wtWx   = (_Float16*)w; w += (size_t)8 * 48 * 512 * 2;
    _Float16* wtWdt  = (_Float16*)w; w += (size_t)8 * 512 * 16 * 2;
    _Float16* wtWout = (_Float16*)w; w += (size_t)8 * 256 * 512 * 2;

    // one-shot weight convert+transpose (all 8 layer-direction sets)
    for (int pd = 0; pd < 8; ++pd) {
        wcvt_kernel<<<(256 * 1024 + 255) / 256, 256, 0, stream>>>(
            Win + (size_t)pd * 256 * 1024, wtWin + (size_t)pd * 1024 * 256, 256, 1024);
        wcvt_kernel<<<(512 * 48 + 255) / 256, 256, 0, stream>>>(
            Wx + (size_t)pd * 512 * 48, wtWx + (size_t)pd * 48 * 512, 512, 48);
        wcvt_kernel<<<(16 * 512 + 255) / 256, 256, 0, stream>>>(
            Wdt + (size_t)pd * 16 * 512, wtWdt + (size_t)pd * 512 * 16, 16, 512);
        wcvt_kernel<<<(512 * 256 + 255) / 256, 256, 0, stream>>>(
            Wout + (size_t)pd * 512 * 256, wtWout + (size_t)pd * 256 * 512, 512, 256);
    }

    inproj_kernel<<<Mrows, 256, 0, stream>>>(x, Wp, bp, h_buf);

    for (int l = 0; l < 4; ++l) {
        ln_kernel<<<Mrows, 256, 0, stream>>>(h_buf, ln_g + l * 256, ln_b + l * 256, hn_buf);
        for (int dir = 0; dir < 2; ++dir) {
            int pd = l * 2 + dir;
            // xz = hn(rev?) @ Win           [M x 256] @ [256 x 1024]
            gemm_wmma<256, 256, 1024, 0, false, false>
                <<<dim3(Mrows / 64, 16), 128, 0, stream>>>(
                hn_buf, wtWin + (size_t)pd * 1024 * 256, nullptr, xz_buf, nullptr, dir, 0);
            // xs = silu(causal_conv(xc))
            conv_silu_kernel<<<(Mrows * 512) / 256, 256, 0, stream>>>(
                xz_buf, convw + (size_t)pd * 512 * 4, convb + (size_t)pd * 512, xs_buf);
            // proj = xs @ Wx                [M x 512] @ [512 x 48]
            gemm_wmma<512, 512, 48, 0, false, false>
                <<<dim3(Mrows / 64, 1), 128, 0, stream>>>(
                xs_buf, wtWx + (size_t)pd * 48 * 512, nullptr, pj_buf, nullptr, 0, 0);
            // dt = softplus(proj[:, :16] @ Wdt + bdt)   [M x 16] @ [16 x 512]
            gemm_wmma<16, 48, 512, 1, true, false>
                <<<dim3(Mrows / 64, 8), 128, 0, stream>>>(
                pj_buf, wtWdt + (size_t)pd * 512 * 16, bdt + (size_t)pd * 512, dt_buf, nullptr, 0, 0);
            // selective scan + gated epilogue -> yz
            scan_kernel<<<(Bsz * 512) / 256, 256, 0, stream>>>(
                dt_buf, xs_buf, pj_buf, xz_buf,
                Alog + (size_t)pd * 512 * 16, Dp + (size_t)pd * 512, yz_buf);
            // h += yz @ Wout (rows reversed back for dir==1)  [M x 512] @ [512 x 256]
            gemm_wmma<512, 512, 256, 0, false, true>
                <<<dim3(Mrows / 64, 4), 128, 0, stream>>>(
                yz_buf, wtWout + (size_t)pd * 256 * 512, nullptr, nullptr, h_buf, 0, dir);
        }
    }

    pool_kernel<<<64, 256, 0, stream>>>(h_buf, pooled);
    head_kernel<<<64, 128, 0, stream>>>(pooled, lnf_g, lnf_b, Wc1, bc1, Wc2, bc2,
                                        (float*)d_out);
}
